// MultiHeadAttention_1632087573277
// MI455X (gfx1250) — matrix operand order repaired and checked by reading, not compiled, not run on hardware
//
#include <hip/hip_runtime.h>

// ---------------------------------------------------------------------------
// MI455X (gfx1250) causal multi-head attention block, wave32 WMMA f16.
//   prep: x,W -> f16 (W transposed [n][k])
//   qkv = x @ W_attn + b_attn  (WMMA, async-LDS double-buffered pipeline)
//   flash attention per head   (WMMA QK^T + PV, online softmax, mask-free bulk,
//                               permlane16 VALU butterfly reductions)
//   out = o @ W_proj + b_proj  (WMMA, same pipeline)
// ---------------------------------------------------------------------------

typedef __attribute__((ext_vector_type(16))) _Float16 v16h;
typedef __attribute__((ext_vector_type(8)))  _Float16 v8h;
typedef __attribute__((ext_vector_type(4)))  _Float16 v4h;
typedef __attribute__((ext_vector_type(8)))  float    v8f;
typedef __attribute__((ext_vector_type(4)))  int      v4i;

#define D_MODEL 1024
#define SEQ     2048
#define NHEAD   16
#define DHEAD   64
#define BH      32

#if defined(__has_builtin)
#if __has_builtin(__builtin_amdgcn_global_load_async_to_lds_b128)
#define HAVE_ASYNC_LDS 1
#endif
#if __has_builtin(__builtin_amdgcn_permlane16)
#define HAVE_PERMLANE16 1
#endif
#endif

#ifdef HAVE_ASYNC_LDS
typedef __attribute__((address_space(1))) v4i g_v4i;
typedef __attribute__((address_space(3))) v4i l_v4i;
__device__ __forceinline__ void async_cp16(const _Float16* g, _Float16* l) {
  __builtin_amdgcn_global_load_async_to_lds_b128(
      (g_v4i*)(void*)g, (l_v4i*)l, 0, 0);
}
template <int N> __device__ __forceinline__ void wait_async() {
  asm volatile("s_wait_asynccnt %0" ::"n"(N) : "memory");
}
#endif

// D = A(16x32) * B(32x16) + C  (f16 in, f32 accum)
__device__ __forceinline__ v8f wmma32(v16h a, v16h b, v8f c) {
  return __builtin_amdgcn_wmma_f32_16x16x32_f16(false, a, false, b,
                                                (short)0, c, false, false);
}

__device__ __forceinline__ v16h ld16c(const _Float16* p) {
  v8h a = *(const v8h*)p;
  v8h b = *(const v8h*)(p + 8);
  v16h r;
#pragma unroll
  for (int i = 0; i < 8; ++i) { r[i] = a[i]; r[i + 8] = b[i]; }
  return r;
}
__device__ __forceinline__ v16h ld8x2(const _Float16* p0, const _Float16* p1) {
  v8h a = *(const v8h*)p0;
  v8h b = *(const v8h*)p1;
  v16h r;
#pragma unroll
  for (int i = 0; i < 8; ++i) { r[i] = a[i]; r[i + 8] = b[i]; }
  return r;
}

// ---- 16-lane butterfly exchange: VALU v_permlane16_b32, no LDS round-trip ----
#ifdef HAVE_PERMLANE16
__device__ constexpr unsigned selpack(int base, int M) {
  unsigned s = 0;
  for (int j = 0; j < 8; ++j)
    s |= ((unsigned)(((base + j) ^ M) & 15)) << (4 * j);
  return s;
}
template <int M> __device__ __forceinline__ float lanexor16(float v) {
  int r = __builtin_amdgcn_permlane16(__float_as_int(v), __float_as_int(v),
                                      selpack(0, M), selpack(8, M),
                                      false, false);
  return __int_as_float(r);
}
#else
template <int M> __device__ __forceinline__ float lanexor16(float v) {
  return __shfl_xor(v, M);
}
#endif

__device__ __forceinline__ float redmax16(float v) {
  v = fmaxf(v, lanexor16<1>(v));
  v = fmaxf(v, lanexor16<2>(v));
  v = fmaxf(v, lanexor16<4>(v));
  v = fmaxf(v, lanexor16<8>(v));
  return v;
}
__device__ __forceinline__ float redsum16(float v) {
  v += lanexor16<1>(v);
  v += lanexor16<2>(v);
  v += lanexor16<4>(v);
  v += lanexor16<8>(v);
  return v;
}

// ---------------------------------------------------------------------------
// Prep kernels (one-shot, bandwidth-trivial): f32 -> f16, W transposed [n][k].
// ---------------------------------------------------------------------------
__global__ __launch_bounds__(256) void cvt_f16(const float* __restrict__ src,
                                               _Float16* __restrict__ dst) {
  int i = (blockIdx.x * 256 + threadIdx.x) * 4;
  float4 v = *(const float4*)(src + i);
  v4h o;
  o[0] = (_Float16)v.x; o[1] = (_Float16)v.y;
  o[2] = (_Float16)v.z; o[3] = (_Float16)v.w;
  *(v4h*)(dst + i) = o;
}

__global__ __launch_bounds__(256) void transpose_cvt(
    const float* __restrict__ W, _Float16* __restrict__ Wt, int N) {
  __shared__ float t[32][33];
  const int n0 = blockIdx.x * 32, k0 = blockIdx.y * 32;
  const int tx = threadIdx.x & 31, ty = threadIdx.x >> 5;  // 32 x 8
#pragma unroll
  for (int i = 0; i < 4; ++i)
    t[ty + i * 8][tx] = W[(size_t)(k0 + ty + i * 8) * N + n0 + tx];
  __syncthreads();
#pragma unroll
  for (int i = 0; i < 4; ++i)
    Wt[(size_t)(n0 + ty + i * 8) * D_MODEL + k0 + tx] =
        (_Float16)t[tx][ty + i * 8];
}

// ---------------------------------------------------------------------------
// WMMA tile compute: 2x4 16x16 tiles per wave from As/Bs [row][32] f16.
// ---------------------------------------------------------------------------
__device__ __forceinline__ void tile_mma(const _Float16* As, const _Float16* Bs,
                                         int wm, int wn, int ln, int kbA,
                                         int kbB, v8f acc[2][4]) {
  v16h a[2], bf[4];
#pragma unroll
  for (int rt = 0; rt < 2; ++rt) {
    const _Float16* p = As + (wm + rt * 16 + ln) * 32 + kbA;
    a[rt] = ld16c(p);
  }
#pragma unroll
  for (int ct = 0; ct < 4; ++ct)
    bf[ct] = ld16c(Bs + (wn + ct * 16 + ln) * 32 + kbB);
#pragma unroll
  for (int rt = 0; rt < 2; ++rt)
#pragma unroll
    for (int ct = 0; ct < 4; ++ct)
      acc[rt][ct] = wmma32(a[rt], bf[ct], acc[rt][ct]);
}

// ---------------------------------------------------------------------------
// GEMM: C = A16[4096][1024] @ Bt^T  (Bt is [N][1024] f16, pre-transposed).
// Block 256 thr (8 waves), tile 128x128, BK=32, double-buffered async LDS.
// MODE 0: scatter into Q (pre-scaled by 1/8) / K / Vt.  MODE 1: f32 out.
// ---------------------------------------------------------------------------
template <int MODE>
__global__ __launch_bounds__(256) void gemm_wmma(
    const _Float16* __restrict__ A16, const _Float16* __restrict__ Bt,
    const float* __restrict__ bias, _Float16* __restrict__ qb,
    _Float16* __restrict__ kbuf, _Float16* __restrict__ vtb,
    float* __restrict__ out) {
  __shared__ __attribute__((aligned(16))) _Float16 As[2][128 * 32];
  __shared__ __attribute__((aligned(16))) _Float16 Bs[2][128 * 32];

  const int tid = threadIdx.x, wid = tid >> 5, lane = tid & 31;
  const int ln = lane & 15, hi = lane >> 4;
  const int bm = blockIdx.y * 128, bn = blockIdx.x * 128;
  const int wm = (wid >> 1) * 32, wn = (wid & 1) * 64;
  const int kbA = hi ? 16 : 0, kbB = hi ? 16 : 0;
  v8f acc[2][4] = {};

#ifdef HAVE_ASYNC_LDS
  // ---- async global->LDS double-buffered pipeline (4 b128 per thr/stage) ----
#pragma unroll
  for (int i = 0; i < 2; ++i) {
    int uu = tid + i * 256, row = uu >> 2, sg = (uu & 3) * 8;
    async_cp16(A16 + (size_t)(bm + row) * D_MODEL + sg, &As[0][row * 32 + sg]);
    async_cp16(Bt + (size_t)(bn + row) * D_MODEL + sg, &Bs[0][row * 32 + sg]);
  }
  for (int s = 0; s < 32; ++s) {
    const int cur = s & 1;
    if (s + 1 < 32) {
      const int k0 = (s + 1) * 32;
#pragma unroll
      for (int i = 0; i < 2; ++i) {
        int uu = tid + i * 256, row = uu >> 2, sg = (uu & 3) * 8;
        async_cp16(A16 + (size_t)(bm + row) * D_MODEL + k0 + sg,
                   &As[1 - cur][row * 32 + sg]);
        async_cp16(Bt + (size_t)(bn + row) * D_MODEL + k0 + sg,
                   &Bs[1 - cur][row * 32 + sg]);
      }
      wait_async<4>();                   // only next stage's copies in flight
    } else {
      wait_async<0>();
    }
    __syncthreads();
    tile_mma(As[cur], Bs[cur], wm, wn, ln, kbA, kbB, acc);
    __syncthreads();
  }
#else
  // ---- fallback: grouped register-staged copies (all loads before stores) ----
  for (int s = 0; s < 32; ++s) {
    const int k0 = s * 32;
    v8h va[2], vb[2];
#pragma unroll
    for (int i = 0; i < 2; ++i) {
      int uu = tid + i * 256, row = uu >> 2, sg = (uu & 3) * 8;
      va[i] = *(const v8h*)(A16 + (size_t)(bm + row) * D_MODEL + k0 + sg);
      vb[i] = *(const v8h*)(Bt + (size_t)(bn + row) * D_MODEL + k0 + sg);
    }
#pragma unroll
    for (int i = 0; i < 2; ++i) {
      int uu = tid + i * 256, row = uu >> 2, sg = (uu & 3) * 8;
      *(v8h*)(&As[0][row * 32 + sg]) = va[i];
      *(v8h*)(&Bs[0][row * 32 + sg]) = vb[i];
    }
    __syncthreads();
    tile_mma(As[0], Bs[0], wm, wn, ln, kbA, kbB, acc);
    __syncthreads();
  }
#endif

  // ---- epilogue ----
#pragma unroll
  for (int rt = 0; rt < 2; ++rt)
#pragma unroll
    for (int ct = 0; ct < 4; ++ct)
#pragma unroll
      for (int r = 0; r < 8; ++r) {
        int m = bm + wm + rt * 16 + r + hi * 8;
        int c = bn + wn + ct * 16 + ln;
        float val = acc[rt][ct][r] + bias[c];
        if (MODE == 1) {
          out[(size_t)m * D_MODEL + c] = val;
        } else {
          int sec = c >> 10, cc = c & 1023;
          int h = cc >> 6, dh = cc & 63;
          int bt = m >> 11, sq = m & 2047, bhh = bt * NHEAD + h;
          if (sec == 0)  // fold softmax scale 1/sqrt(64) into Q
            qb[((size_t)bhh * SEQ + sq) * DHEAD + dh] = (_Float16)(val * 0.125f);
          else if (sec == 1)
            kbuf[((size_t)bhh * SEQ + sq) * DHEAD + dh] = (_Float16)val;
          else
            vtb[((size_t)bhh * DHEAD + dh) * SEQ + sq] = (_Float16)val;
        }
      }
}

// ---------------------------------------------------------------------------
// One 32-key flash-attention step for one wave (16 queries).
// MASK=false for fully-causal-legal blocks (no compares at all),
// MASK=true only for the single diagonal block.
// ---------------------------------------------------------------------------
template <bool MASK>
__device__ __forceinline__ void attn_step(
    int kblk, int Qw, int ln, int hi, int kbA, int kbB,
    const _Float16* __restrict__ kb_bh, const _Float16* __restrict__ vt_bh,
    _Float16* __restrict__ pw, const v16h& qa0, const v16h& qa1,
    v8f oacc[4], float mrun[8], float lrun[8]) {
  const _Float16* kp0 = kb_bh + (size_t)(kblk + ln) * DHEAD;
  const _Float16* kp1 = kb_bh + (size_t)(kblk + 16 + ln) * DHEAD;
  v8f s0 = {}, s1 = {};
  s0 = wmma32(qa0, ld16c(kp0 + kbB), s0);
  s0 = wmma32(qa1, ld16c(kp0 + 32 + kbB), s0);
  s1 = wmma32(qa0, ld16c(kp1 + kbB), s1);
  s1 = wmma32(qa1, ld16c(kp1 + 32 + kbB), s1);

  float p0v[8], p1v[8];
#pragma unroll
  for (int r = 0; r < 8; ++r) {
    float a0 = s0[r], a1 = s1[r];     // Q pre-scaled by 1/sqrt(64)
    if (MASK) {
      int qq = Qw + r + hi * 8;
      a0 = (kblk + ln      > qq) ? -3.0e38f : a0;
      a1 = (kblk + 16 + ln > qq) ? -3.0e38f : a1;
    }
    float rm = redmax16(fmaxf(a0, a1));
    float mn = fmaxf(mrun[r], rm);
    float sc = __expf(mrun[r] - mn);
    float e0 = __expf(a0 - mn);
    float e1 = __expf(a1 - mn);
    float rs = redsum16(e0 + e1);
    lrun[r] = lrun[r] * sc + rs;
    mrun[r] = mn;
    oacc[0][r] *= sc; oacc[1][r] *= sc; oacc[2][r] *= sc; oacc[3][r] *= sc;
    p0v[r] = e0; p1v[r] = e1;
  }

  // restage P (C/D layout) -> A-fragment layout through per-wave LDS
#pragma unroll
  for (int r = 0; r < 8; ++r) {
    int M = r + hi * 8;
    pw[M * 32 + ln]      = (_Float16)p0v[r];
    pw[M * 32 + 16 + ln] = (_Float16)p1v[r];
  }
  asm volatile("s_wait_dscnt 0" ::: "memory");  // intra-wave DS ordering
  const _Float16* pr = pw + ln * 32 + kbA;
  v16h pa = ld16c(pr);

  const _Float16* vp = vt_bh + kblk + kbB;
#pragma unroll
  for (int nt = 0; nt < 4; ++nt) {
    v16h vb = ld16c(vp + (size_t)(nt * 16 + ln) * SEQ);
    oacc[nt] = wmma32(pa, vb, oacc[nt]);
  }
}

// ---------------------------------------------------------------------------
// Flash attention. Grid (q_tiles=16, bh=32), 256 thr = 8 waves; each wave owns
// 16 queries; Qw>>5 mask-free 32-key blocks + exactly one diagonal block.
// ---------------------------------------------------------------------------
__global__ __launch_bounds__(256) void attn_kernel(
    const _Float16* __restrict__ qb, const _Float16* __restrict__ kb,
    const _Float16* __restrict__ vtb, _Float16* __restrict__ ob) {
  __shared__ __attribute__((aligned(16))) _Float16 plds[8][16 * 32];

  const int tid = threadIdx.x, wid = tid >> 5, lane = tid & 31;
  const int ln = lane & 15, hi = lane >> 4;
  const int bh = blockIdx.y;
  const int bt = bh >> 4, h = bh & 15;
  const int Qw = blockIdx.x * 128 + wid * 16;
  const int kbA = hi ? 16 : 0, kbB = hi ? 16 : 0;

  const _Float16* kb_bh = kb + (size_t)bh * SEQ * DHEAD;
  const _Float16* vt_bh = vtb + (size_t)bh * DHEAD * SEQ;
  _Float16* pw = plds[wid];

  const _Float16* qp = qb + ((size_t)bh * SEQ + Qw + ln) * DHEAD;
  const v16h qa0 = ld16c(qp + kbA);
  const v16h qa1 = ld16c(qp + 32 + kbA);

  v8f oacc[4] = {};
  float mrun[8], lrun[8];
#pragma unroll
  for (int r = 0; r < 8; ++r) { mrun[r] = -3.0e38f; lrun[r] = 0.0f; }

  const int nfull = Qw >> 5;                       // mask-free blocks
  for (int b = 0; b < nfull; ++b)
    attn_step<false>(b * 32, Qw, ln, hi, kbA, kbB, kb_bh, vt_bh, pw,
                     qa0, qa1, oacc, mrun, lrun);
  attn_step<true>(nfull * 32, Qw, ln, hi, kbA, kbB, kb_bh, vt_bh, pw,
                  qa0, qa1, oacc, mrun, lrun);     // diagonal block

#pragma unroll
  for (int r = 0; r < 8; ++r) {
    float inv = 1.0f / lrun[r];
    int qq = Qw + r + hi * 8;
    _Float16* orow = ob + ((size_t)bt * SEQ + qq) * D_MODEL + h * DHEAD;
#pragma unroll
    for (int nt = 0; nt < 4; ++nt)
      orow[nt * 16 + ln] = (_Float16)(oacc[nt][r] * inv);
  }
}

// ---------------------------------------------------------------------------
extern "C" void kernel_launch(void* const* d_in, const int* in_sizes, int n_in,
                              void* d_out, int out_size, void* d_ws, size_t ws_size,
                              hipStream_t stream) {
  (void)in_sizes; (void)n_in; (void)out_size; (void)ws_size;
  const float* x      = (const float*)d_in[0];
  const float* W_attn = (const float*)d_in[1];
  const float* b_attn = (const float*)d_in[2];
  const float* W_proj = (const float*)d_in[3];
  const float* b_proj = (const float*)d_in[4];
  float* out = (float*)d_out;

  const size_t per = (size_t)BH * SEQ * DHEAD;  // 4M halfs
  _Float16* q   = (_Float16*)d_ws;
  _Float16* k   = q + per;
  _Float16* vt  = k + per;
  _Float16* o   = vt + per;                     // [4096][1024] f16
  _Float16* x16 = o + (size_t)4096 * D_MODEL;   // [4096][1024] f16
  _Float16* wat = x16 + (size_t)4096 * D_MODEL; // [3072][1024] f16 (transposed)
  _Float16* wpt = wat + (size_t)3072 * D_MODEL; // [1024][1024] f16 (transposed)

  cvt_f16<<<4096, 256, 0, stream>>>(x, x16);
  transpose_cvt<<<dim3(96, 32), 256, 0, stream>>>(W_attn, wat, 3072);
  transpose_cvt<<<dim3(32, 32), 256, 0, stream>>>(W_proj, wpt, 1024);

  gemm_wmma<0><<<dim3(24, 32), 256, 0, stream>>>(x16, wat, b_attn, q, k, vt,
                                                 nullptr);
  attn_kernel<<<dim3(16, 32), 256, 0, stream>>>(q, k, vt, o);
  gemm_wmma<1><<<dim3(8, 32), 256, 0, stream>>>(o, wpt, b_proj, nullptr,
                                                nullptr, nullptr, out);
}
